// LinearTransformerBlock_4440996184662
// MI455X (gfx1250) — matrix operand order repaired and checked by reading, not compiled, not run on hardware
//
#include <hip/hip_runtime.h>
#include <hip/hip_bf16.h>

// Problem dims (fixed by reference)
#define B_    4
#define T_    8192
#define H_    16
#define D_    64
#define C_    64                   // chunk length
#define NSEG  8                    // parallel segments over T
#define SEGT  (T_ / NSEG)          // 1024
#define NCH_SEG (SEGT / C_)        // 16 chunks per segment
#define LDB   72                   // bf16 LDS row stride (pad; 144B = 16B aligned)
#define NA    80                   // augmented N dim (64 + den col + pad), 5 tiles
#define STATE_SZ 4160              // per (bh,seg) state floats: 64x64 S + 64 z
#define EPSF  1e-6f

typedef __attribute__((ext_vector_type(16))) __bf16 v16bf;
typedef __attribute__((ext_vector_type(8)))  float  v8f;

union BFRAG { v16bf v; unsigned short u[16]; };

__device__ __forceinline__ unsigned short f2bf(float f) {   // hardware cvt
  union { __bf16 b; unsigned short u; } x;
  x.b = (__bf16)f;
  return x.u;
}
__device__ __forceinline__ float phi(float x) {   // elu(x)+1
  return (x > 0.f) ? (x + 1.f) : __expf(x);
}
// K position inside the v16bf A-fragment for element e, given lane half
__device__ __forceinline__ int a_kloc(int half, int e) {
  int vr = e >> 1, j = e & 1;
  return (vr < 4) ? (half * 8 + 2 * vr + j) : (16 + half * 8 + 2 * (vr - 4) + j);
}
__device__ __forceinline__ v8f wmma_bf16(v16bf a, v16bf b, v8f c) {
  return __builtin_amdgcn_wmma_f32_16x16x32_bf16(false, a, false, b, (short)0, c,
                                                 false, false);
}
// A-fragment from a row-major bf16 row pointer (32B contiguous -> 2x b128)
__device__ __forceinline__ v16bf ldA(const unsigned short* row, int kk, int half) {
  BFRAG a;
#pragma unroll
  for (int e = 0; e < 16; ++e) a.u[e] = row[kk + half * 16 + e];
  return a.v;
}
// B-fragment from an N-major (transposed) bf16 row pointer (32B contiguous -> 2x b128)
__device__ __forceinline__ v16bf ldB(const unsigned short* row, int kk, int half) {
  BFRAG b;
#pragma unroll
  for (int e = 0; e < 16; ++e) b.u[e] = row[kk + half * 16 + e];
  return b.v;
}
// One state-tile update: S'(tm,tn) += phiK^T * V'  (A from sKT rows, B from sVT rows)
__device__ __forceinline__ v8f state_upd(v8f acc, const unsigned short* sKT_,
                                         const unsigned short* sVT_,
                                         int tm, int tn, int half, int mloc) {
  const unsigned short* ar = sKT_ + (tm * 16 + mloc) * LDB;
  const unsigned short* br = sVT_ + (tn * 16 + mloc) * LDB;
#pragma unroll
  for (int kk = 0; kk < C_; kk += 32)
    acc = wmma_bf16(ldA(ar, kk, half), ldB(br, kk, half), acc);
  return acc;
}
// One output tile: A_masked*V' + phiQ*S'
__device__ __forceinline__ v8f out_tile(const unsigned short* sA_, const unsigned short* sQ_,
                                        const unsigned short* sVT_, const unsigned short* sST_,
                                        int tm, int tn, int half, int mloc) {
  v8f acc = {};
  const unsigned short* a1 = sA_ + (tm * 16 + mloc) * LDB;
  const unsigned short* b1 = sVT_ + (tn * 16 + mloc) * LDB;
#pragma unroll
  for (int kk = 0; kk < C_; kk += 32)
    acc = wmma_bf16(ldA(a1, kk, half), ldB(b1, kk, half), acc);
  const unsigned short* a2 = sQ_ + (tm * 16 + mloc) * LDB;
  const unsigned short* b2 = sST_ + (tn * 16 + mloc) * LDB;
#pragma unroll
  for (int kk = 0; kk < D_; kk += 32)
    acc = wmma_bf16(ldA(a2, kk, half), ldB(b2, kk, half), acc);
  return acc;
}
__device__ __forceinline__ void store_state(float* wsb, v8f acc, int tm, int tn,
                                            int half, int mloc) {
#pragma unroll
  for (int vr = 0; vr < 8; ++vr) {
    int m = tm * 16 + vr + half * 8;
    if (tn < 4) wsb[m * 64 + tn * 16 + mloc] = acc[vr];
    else if (mloc == 0) wsb[4096 + m] = acc[vr];
  }
}
__device__ __forceinline__ v8f load_state(const float* wsb, int tm, int tn,
                                          int half, int mloc) {
  v8f a;
#pragma unroll
  for (int vr = 0; vr < 8; ++vr) {
    int m = tm * 16 + vr + half * 8;
    a[vr] = (tn < 4) ? wsb[m * 64 + tn * 16 + mloc]
                     : ((mloc == 0) ? wsb[4096 + m] : 0.f);
  }
  return a;
}
__device__ __forceinline__ void stash_sst(unsigned short* sST_, v8f acc, int tm, int tn,
                                          int half, int mloc) {
#pragma unroll
  for (int vr = 0; vr < 8; ++vr) {
    int m = tm * 16 + vr + half * 8;
    if (tn < 4) sST_[(tn * 16 + mloc) * LDB + m] = f2bf(acc[vr]);
    else if (mloc == 0) sST_[64 * LDB + m] = f2bf(acc[vr]);
  }
}

// ---------------- K1: per-segment inclusive state sums S'_seg = phiK^T V' -------------
__global__ __launch_bounds__(256, 1)
void k1_segstate(const float* __restrict__ k, const float* __restrict__ v,
                 float* __restrict__ ws) {
  __shared__ unsigned short sKT[D_ * LDB];   // phiK transposed (d-major)
  __shared__ unsigned short sVT[NA * LDB];   // V' transposed (n-major), row64=1, 65..79=0

  const int tid = threadIdx.x, lane = tid & 31, wave = tid >> 5;
  const int half = lane >> 4, mloc = lane & 15;
  const int blk = blockIdx.x, bh = blk / NSEG, seg = blk % NSEG;
  const int b = bh / H_, h = bh % H_;
  const size_t base = (size_t)b * T_ * (H_ * D_) + (size_t)h * D_;

  for (int i = tid; i < 16 * LDB; i += 256) {      // constant rows of V'
    int rr = i / LDB;
    sVT[(64 + rr) * LDB + (i % LDB)] = (rr == 0) ? 0x3F80 : 0;
  }
  v8f S0 = {}, S1 = {}, S2 = {};
  const int t0seg = seg * SEGT;
  for (int c = 0; c < NCH_SEG; ++c) {
    const int t0 = t0seg + c * C_;
    {
      const int r = tid >> 2, d0 = (tid & 3) * 16;
      const float* kp = k + base + (size_t)(t0 + r) * (H_ * D_) + d0;
      const float* vp = v + base + (size_t)(t0 + r) * (H_ * D_) + d0;
      if (c + 1 < NCH_SEG) {
        __builtin_prefetch(kp + (size_t)C_ * (H_ * D_), 0, 1);
        __builtin_prefetch(vp + (size_t)C_ * (H_ * D_), 0, 1);
      }
#pragma unroll
      for (int i = 0; i < 16; i += 4) {
        float4 kv = *(const float4*)(kp + i);
        float4 vv = *(const float4*)(vp + i);
        sKT[(d0 + i + 0) * LDB + r] = f2bf(phi(kv.x));
        sKT[(d0 + i + 1) * LDB + r] = f2bf(phi(kv.y));
        sKT[(d0 + i + 2) * LDB + r] = f2bf(phi(kv.z));
        sKT[(d0 + i + 3) * LDB + r] = f2bf(phi(kv.w));
        sVT[(d0 + i + 0) * LDB + r] = f2bf(vv.x);
        sVT[(d0 + i + 1) * LDB + r] = f2bf(vv.y);
        sVT[(d0 + i + 2) * LDB + r] = f2bf(vv.z);
        sVT[(d0 + i + 3) * LDB + r] = f2bf(vv.w);
      }
    }
    __syncthreads();
    S0 = state_upd(S0, sKT, sVT, wave & 3, wave >> 2, half, mloc);
    S1 = state_upd(S1, sKT, sVT, wave & 3, 2 + (wave >> 2), half, mloc);
    if (wave < 4) S2 = state_upd(S2, sKT, sVT, wave, 4, half, mloc);
    __syncthreads();
  }
  float* wsb = ws + (size_t)blk * STATE_SZ;
  store_state(wsb, S0, wave & 3, wave >> 2, half, mloc);
  store_state(wsb, S1, wave & 3, 2 + (wave >> 2), half, mloc);
  if (wave < 4) store_state(wsb, S2, wave, 4, half, mloc);
}

// ---------------- K2: exclusive scan of segment states per (b,h) --------------------
__global__ __launch_bounds__(256, 1)
void k2_scan(float* __restrict__ ws) {
  __shared__ float4 run[STATE_SZ / 4];
  const int tid = threadIdx.x;
  for (int i = tid; i < STATE_SZ / 4; i += 256)
    run[i] = make_float4(0.f, 0.f, 0.f, 0.f);
  float4* base = (float4*)(ws + (size_t)blockIdx.x * NSEG * STATE_SZ);
  for (int s = 0; s < NSEG; ++s) {
    float4* p = base + (size_t)s * (STATE_SZ / 4);
    for (int i = tid; i < STATE_SZ / 4; i += 256) {
      float4 t = p[i], r = run[i];
      p[i] = r;
      r.x += t.x; r.y += t.y; r.z += t.z; r.w += t.w;
      run[i] = r;
    }
  }
}

// ---------------- K3: chunked linear attention within each segment ------------------
__global__ __launch_bounds__(256, 1)
void k3_attn(const float* __restrict__ q, const float* __restrict__ k,
             const float* __restrict__ v, const float* __restrict__ ws,
             float* __restrict__ out) {
  __shared__ unsigned short sQ[C_ * LDB];    // phiQ row-major
  __shared__ unsigned short sK[C_ * LDB];    // phiK row-major (B of QK^T)
  __shared__ unsigned short sKT[D_ * LDB];   // phiK transposed (A of K^T V')
  __shared__ unsigned short sA[C_ * LDB];    // masked attention
  __shared__ unsigned short sVT[NA * LDB];   // V' transposed
  __shared__ unsigned short sST[NA * LDB];   // S' transposed (bf16 mirror)
  __shared__ float sDen[C_];

  const int tid = threadIdx.x, lane = tid & 31, wave = tid >> 5;
  const int half = lane >> 4, mloc = lane & 15;
  const int blk = blockIdx.x, bh = blk / NSEG, seg = blk % NSEG;
  const int b = bh / H_, h = bh % H_;
  const size_t base = (size_t)b * T_ * (H_ * D_) + (size_t)h * D_;
  const float* wsb = ws + (size_t)blk * STATE_SZ;

  // init transposed state mirror + constant V' rows
  for (int i = tid; i < NA * LDB; i += 256) sST[i] = 0;
  for (int i = tid; i < 16 * LDB; i += 256) {
    int rr = i / LDB;
    sVT[(64 + rr) * LDB + (i % LDB)] = (rr == 0) ? 0x3F80 : 0;
  }
  __syncthreads();
  for (int i = tid; i < 4096; i += 256) {
    int m = i >> 6, n = i & 63;
    sST[n * LDB + m] = f2bf(wsb[i]);
  }
  if (tid < D_) sST[64 * LDB + tid] = f2bf(wsb[4096 + tid]);

  // persistent state registers (20 tiles over 8 waves)
  v8f S0 = load_state(wsb, wave & 3, wave >> 2, half, mloc);
  v8f S1 = load_state(wsb, wave & 3, 2 + (wave >> 2), half, mloc);
  v8f S2 = {};
  if (wave < 4) S2 = load_state(wsb, wave, 4, half, mloc);
  __syncthreads();

  const int t0seg = seg * SEGT;
  for (int c = 0; c < NCH_SEG; ++c) {
    const int t0 = t0seg + c * C_;
    // ---- load + featurize ----
    {
      const int r = tid >> 2, d0 = (tid & 3) * 16;
      const float* qp = q + base + (size_t)(t0 + r) * (H_ * D_) + d0;
      const float* kp = k + base + (size_t)(t0 + r) * (H_ * D_) + d0;
      const float* vp = v + base + (size_t)(t0 + r) * (H_ * D_) + d0;
      if (c + 1 < NCH_SEG) {
        __builtin_prefetch(qp + (size_t)C_ * (H_ * D_), 0, 1);
        __builtin_prefetch(kp + (size_t)C_ * (H_ * D_), 0, 1);
        __builtin_prefetch(vp + (size_t)C_ * (H_ * D_), 0, 1);
      }
      unsigned short* qd = &sQ[r * LDB + d0];
      unsigned short* kd = &sK[r * LDB + d0];
#pragma unroll
      for (int i = 0; i < 16; i += 4) {
        float4 qv = *(const float4*)(qp + i);
        float4 kv = *(const float4*)(kp + i);
        float4 vv = *(const float4*)(vp + i);
        qd[i + 0] = f2bf(phi(qv.x)); qd[i + 1] = f2bf(phi(qv.y));
        qd[i + 2] = f2bf(phi(qv.z)); qd[i + 3] = f2bf(phi(qv.w));
        unsigned short k0 = f2bf(phi(kv.x)), k1 = f2bf(phi(kv.y));
        unsigned short k2 = f2bf(phi(kv.z)), k3 = f2bf(phi(kv.w));
        kd[i + 0] = k0; kd[i + 1] = k1; kd[i + 2] = k2; kd[i + 3] = k3;
        sKT[(d0 + i + 0) * LDB + r] = k0; sKT[(d0 + i + 1) * LDB + r] = k1;
        sKT[(d0 + i + 2) * LDB + r] = k2; sKT[(d0 + i + 3) * LDB + r] = k3;
        sVT[(d0 + i + 0) * LDB + r] = f2bf(vv.x);
        sVT[(d0 + i + 1) * LDB + r] = f2bf(vv.y);
        sVT[(d0 + i + 2) * LDB + r] = f2bf(vv.z);
        sVT[(d0 + i + 3) * LDB + r] = f2bf(vv.w);
      }
    }
    __syncthreads();

    // ---- phase A: A = phiQ * phiK^T, causal mask, spill bf16 ----
#pragma unroll
    for (int t = 0; t < 2; ++t) {
      const int ti = 2 * wave + t, tm = ti >> 2, tn = ti & 3;
      v8f acc = {};
      const unsigned short* ar = &sQ[(tm * 16 + mloc) * LDB];
      const unsigned short* br = &sK[(tn * 16 + mloc) * LDB];
#pragma unroll
      for (int kk = 0; kk < D_; kk += 32)
        acc = wmma_bf16(ldA(ar, kk, half), ldB(br, kk, half), acc);
#pragma unroll
      for (int vr = 0; vr < 8; ++vr) {
        const int i = tm * 16 + vr + half * 8, j = tn * 16 + mloc;
        sA[i * LDB + j] = (j <= i) ? f2bf(acc[vr]) : (unsigned short)0;
      }
    }
    __syncthreads();

    // ---- phase O-den: tiles (tm,4) give den = rowsum(A) + q.z ----
    if (wave < 4) {
      v8f acc = out_tile(sA, sQ, sVT, sST, wave, 4, half, mloc);
      if (mloc == 0) {
#pragma unroll
        for (int vr = 0; vr < 8; ++vr) sDen[wave * 16 + vr + half * 8] = acc[vr] + EPSF;
      }
    }
    __syncthreads();

    // ---- phase O-num: O = A*V + phiQ*S ; divide ; store ----
#pragma unroll
    for (int t = 0; t < 2; ++t) {
      const int ti = 2 * wave + t, tm = ti >> 2, tn = ti & 3;
      v8f acc = out_tile(sA, sQ, sVT, sST, tm, tn, half, mloc);
#pragma unroll
      for (int vr = 0; vr < 8; ++vr) {
        const int i = tm * 16 + vr + half * 8;
        out[base + (size_t)(t0 + i) * (H_ * D_) + tn * 16 + mloc] = acc[vr] / sDen[i];
      }
    }
    __syncthreads();

    // ---- phase S: S' += phiK^T * V' (registers) ; refresh bf16 mirror ----
    S0 = state_upd(S0, sKT, sVT, wave & 3, wave >> 2, half, mloc);
    stash_sst(sST, S0, wave & 3, wave >> 2, half, mloc);
    S1 = state_upd(S1, sKT, sVT, wave & 3, 2 + (wave >> 2), half, mloc);
    stash_sst(sST, S1, wave & 3, 2 + (wave >> 2), half, mloc);
    if (wave < 4) {
      S2 = state_upd(S2, sKT, sVT, wave, 4, half, mloc);
      stash_sst(sST, S2, wave, 4, half, mloc);
    }
    __syncthreads();
  }
}

extern "C" void kernel_launch(void* const* d_in, const int* in_sizes, int n_in,
                              void* d_out, int out_size, void* d_ws, size_t ws_size,
                              hipStream_t stream) {
  (void)in_sizes; (void)n_in; (void)out_size; (void)ws_size;
  const float* q = (const float*)d_in[0];
  const float* k = (const float*)d_in[1];
  const float* v = (const float*)d_in[2];
  float* out = (float*)d_out;
  float* ws  = (float*)d_ws;     // needs 64*8*4160*4 B ~= 8.1 MB

  k1_segstate<<<dim3(B_ * H_ * NSEG), dim3(256), 0, stream>>>(k, v, ws);
  k2_scan<<<dim3(B_ * H_), dim3(256), 0, stream>>>(ws);
  k3_attn<<<dim3(B_ * H_ * NSEG), dim3(256), 0, stream>>>(q, k, v, ws, out);
}
